// AttentionLayer_82411832476313
// MI455X (gfx1250) — matrix operand order repaired and checked by reading, not compiled, not run on hardware
//
#include <hip/hip_runtime.h>
#include <hip/hip_bf16.h>

// ---------------------------------------------------------------------------
// AttentionLayer for MI455X (gfx1250).
// All GEMMs via v_wmma_f32_16x16x32_bf16 (wave32, fp32 accumulate).
//
// Roofline: ~70 GFLOP compute; HBM traffic dominated by the attention matrix
// A (536 MB, larger than the 192 MB L2). Fusing scores+softmax+A@V into one
// kernel (16x2048 score strip = 128 KB, fits CDNA5's 320 KB/WGP LDS) means A
// is written to HBM exactly once: ~0.6 GB total traffic (~26 us at 23.3 TB/s)
// instead of ~2.1 GB for the unfused 3-pass version. Everything else
// (inputs/weights/intermediates ~60 MB) is L2-resident.
// ---------------------------------------------------------------------------

#define D_MODEL  1024
#define N_HEADS  16
#define HEAD_DIM 64
#define SEQ      2048
#define BATCH    2
// LDS score strip row stride (floats): 2048 + 4 so rows differing by 8
// (the two half-wave M groups of the WMMA C/D layout) land 32 banks apart.
#define SC_STRIDE 2052

typedef __attribute__((ext_vector_type(16))) __bf16 v16bf;
typedef __attribute__((ext_vector_type(8)))  float  v8f;

// ---------------------------------------------------------------------------
// Fragment loaders per CDNA5 ISA 7.12.2 (wave32):
//   A (16x32 bf16, MxK): lanes 0-15 -> M=lane, lanes 16-31 -> M=lane-16.
//     VGPR0..3 hold K = h*8 + {0..7}, VGPR4..7 hold K = 16 + h*8 + {0..7},
//     h = lane>>4; each VGPR packs 2 consecutive K (lo,hi).
//   B (32x16 bf16, KxN) fed from transposed ("Bt" = [N][K]) storage: lane
//     n = lane&15, elements 0..7 are K = k0 + h*8 + {0..7}, elements 8..15
//     are K = k0 + 16 + h*8 + {0..7} (same K order as A), h = lane>>4.
//   C/D (16x16 f32): element r of lane l -> row m = r + (l>>4)*8, col n=l&15.
// ---------------------------------------------------------------------------

__device__ __forceinline__ v16bf load_a_frag_bf16(const __bf16* __restrict__ base,
                                                  int lda, int m0, int k0) {
  const int lane = threadIdx.x & 31;
  const int h    = lane >> 4;
  const __bf16* row = base + (size_t)(m0 + (lane & 15)) * lda + k0;
  v16bf a;
#pragma unroll
  for (int i = 0; i < 8; ++i) {
    a[i]     = row[h * 8 + i];
    a[i + 8] = row[16 + h * 8 + i];
  }
  return a;
}

// A fragment sourced from an f32 LDS score strip (row stride SC_STRIDE).
__device__ __forceinline__ v16bf load_a_frag_lds_f32(const float* sc, int k0) {
  const int lane = threadIdx.x & 31;
  const int h    = lane >> 4;
  const float* row = sc + (size_t)(lane & 15) * SC_STRIDE + k0;
  v16bf a;
#pragma unroll
  for (int i = 0; i < 8; ++i) {
    a[i]     = (__bf16)row[h * 8 + i];
    a[i + 8] = (__bf16)row[16 + h * 8 + i];
  }
  return a;
}

__device__ __forceinline__ v16bf load_b_frag_bf16(const __bf16* __restrict__ bt,
                                                  int ldb, int n0, int k0) {
  const int lane = threadIdx.x & 31;
  const int h = lane >> 4;
  const __bf16* row = bt + (size_t)(n0 + (lane & 15)) * ldb + k0;
  v16bf b;
#pragma unroll
  for (int i = 0; i < 8; ++i) {
    b[i]     = row[h * 8 + i];
    b[i + 8] = row[16 + h * 8 + i];
  }
  return b;
}

// ---------------------------------------------------------------------------
// Elementwise helpers
// ---------------------------------------------------------------------------
__global__ void f32_to_bf16_kernel(const float* __restrict__ x,
                                   __bf16* __restrict__ y, size_t n) {
  size_t i = (size_t)blockIdx.x * blockDim.x + threadIdx.x;
  if (i < n) y[i] = (__bf16)x[i];
}

// W [K=1024][N=1024] f32  ->  Wt [N][K] bf16
__global__ void transpose_w_kernel(const float* __restrict__ W,
                                   __bf16* __restrict__ Wt) {
  size_t idx = (size_t)blockIdx.x * blockDim.x + threadIdx.x;
  if (idx >= (size_t)D_MODEL * D_MODEL) return;
  int n = (int)(idx % D_MODEL);
  int k = (int)(idx / D_MODEL);
  Wt[(size_t)n * D_MODEL + k] = (__bf16)W[idx];
}

// v bf16 [B][S][H*E] -> Vt bf16 [B][H][E][S]
__global__ void transpose_v_kernel(const __bf16* __restrict__ v,
                                   __bf16* __restrict__ Vt) {
  size_t idx = (size_t)blockIdx.x * blockDim.x + threadIdx.x;
  if (idx >= (size_t)BATCH * SEQ * D_MODEL) return;
  int e  = (int)(idx % HEAD_DIM);
  int hd = (int)((idx / HEAD_DIM) % N_HEADS);
  int s  = (int)((idx / D_MODEL) % SEQ);
  int b  = (int)(idx / ((size_t)SEQ * D_MODEL));
  Vt[(((size_t)b * N_HEADS + hd) * HEAD_DIM + e) * SEQ + s] = v[idx];
}

// ---------------------------------------------------------------------------
// GEMM: Y[M][N] = X[M][K](bf16) @ Wt[N][K]^T(bf16) + bias[N], f32 accumulate.
// Block = 4 waves; each wave owns a 16x64 strip (4 WMMA accumulators reusing
// one A fragment per K-step). grid = (N/256, M/16). Weights + activations are
// L2-resident (192 MB), so global fragment loads are cheap.
// ---------------------------------------------------------------------------
template <typename OutT>
__global__ void gemm_bf16_kernel(const __bf16* __restrict__ X,
                                 const __bf16* __restrict__ Wt,
                                 const float*  __restrict__ bias,
                                 OutT* __restrict__ Y,
                                 int M, int N, int K) {
  const int wave = threadIdx.x >> 5;
  const int lane = threadIdx.x & 31;
  const int m0 = blockIdx.y * 16;
  const int n0 = blockIdx.x * 256 + wave * 64;

  v8f c[4] = {};
  for (int k0 = 0; k0 < K; k0 += 32) {
    v16bf a = load_a_frag_bf16(X, K, m0, k0);
#pragma unroll
    for (int t = 0; t < 4; ++t) {
      v16bf b = load_b_frag_bf16(Wt, K, n0 + t * 16, k0);
      c[t] = __builtin_amdgcn_wmma_f32_16x16x32_bf16(false, a, false, b,
                                                     (short)0, c[t], false, false);
    }
  }

  const int h  = lane >> 4;
  const int nl = lane & 15;
#pragma unroll
  for (int t = 0; t < 4; ++t) {
    const int n = n0 + t * 16 + nl;
    const float bv = bias[n];
#pragma unroll
    for (int r = 0; r < 8; ++r) {
      const int m = m0 + h * 8 + r;
      Y[(size_t)m * N + n] = (OutT)(c[t][r] + bv);
    }
  }
}

// ---------------------------------------------------------------------------
// Fused attention: one block per (b, h, 16-row l-tile); 4 waves / 128 threads.
//   Phase 1: scores  S = 0.125*(tau * q.k + delta) -> LDS (16 x 2048 f32)
//   Phase 2: softmax per row in LDS, normalized A row written to HBM once
//   Phase 3: attn tile = P @ V from LDS fragments (f32 -> bf16 on the fly)
// LDS: 16 * 2052 * 4 B = 131,328 B  (fits the 320 KB/WGP CDNA5 LDS; two
// blocks can still co-reside on a WGP).
// grid = (L/16, B*H).
// ---------------------------------------------------------------------------
__global__ void __launch_bounds__(128)
attn_fused_kernel(const __bf16* __restrict__ qbf,   // [B][L][H*E]
                  const __bf16* __restrict__ kbf,   // [B][S][H*E]
                  const __bf16* __restrict__ Vt,    // [B][H][E][S]
                  const float* __restrict__ tau,    // [B]
                  const float* __restrict__ delta,  // [B][S]
                  float* __restrict__ A,            // [B][H][L][S]
                  __bf16* __restrict__ attn) {      // [B][L][H*E]
  __shared__ float sc[16 * SC_STRIDE];

  const int wave = threadIdx.x >> 5;  // 0..3
  const int lane = threadIdx.x & 31;
  const int h    = lane >> 4;
  const int nl   = lane & 15;
  const int bh = blockIdx.y;
  const int b  = bh / N_HEADS;
  const int hd = bh % N_HEADS;
  const int l0 = blockIdx.x * 16;

  const __bf16* qbase = qbf + (size_t)b * SEQ * D_MODEL + hd * HEAD_DIM;
  const __bf16* kbase = kbf + (size_t)b * SEQ * D_MODEL + hd * HEAD_DIM;

  // ---- Phase 1: scores into LDS ------------------------------------------
  // Q fragments for this l-tile are loop-invariant: load once.
  const v16bf a0 = load_a_frag_bf16(qbase, D_MODEL, l0, 0);
  const v16bf a1 = load_a_frag_bf16(qbase, D_MODEL, l0, 32);
  const float tb = tau[b];
  const float scale = 0.125f;  // 1/sqrt(64)

  for (int st = wave; st < SEQ / 16; st += 4) {
    const int s0 = st * 16;
    v8f c = {};
    v16bf b0 = load_b_frag_bf16(kbase, D_MODEL, s0, 0);   // row s of K = col of K^T
    c = __builtin_amdgcn_wmma_f32_16x16x32_bf16(false, a0, false, b0,
                                                (short)0, c, false, false);
    v16bf b1 = load_b_frag_bf16(kbase, D_MODEL, s0, 32);
    c = __builtin_amdgcn_wmma_f32_16x16x32_bf16(false, a1, false, b1,
                                                (short)0, c, false, false);
    const int s = s0 + nl;
    const float dl = delta[(size_t)b * SEQ + s];
#pragma unroll
    for (int r = 0; r < 8; ++r) {
      sc[(size_t)(h * 8 + r) * SC_STRIDE + s] = scale * (tb * c[r] + dl);
    }
  }
  __syncthreads();

  // ---- Phase 2: per-row softmax in LDS, single HBM write of A ------------
  float* Aout = A + ((size_t)bh * SEQ + l0) * SEQ;
#pragma unroll
  for (int j = 0; j < 4; ++j) {
    const int r = wave * 4 + j;         // 4 waves x 4 rows = 16 rows
    float* row = sc + (size_t)r * SC_STRIDE;

    float m = -3.0e38f;
    for (int i = 0; i < SEQ / 32; ++i) m = fmaxf(m, row[lane + i * 32]);
#pragma unroll
    for (int off = 16; off > 0; off >>= 1) m = fmaxf(m, __shfl_xor(m, off, 32));

    float sum = 0.f;
    for (int i = 0; i < SEQ / 32; ++i) {
      const float e = __expf(row[lane + i * 32] - m);
      row[lane + i * 32] = e;
      sum += e;
    }
#pragma unroll
    for (int off = 16; off > 0; off >>= 1) sum += __shfl_xor(sum, off, 32);

    const float inv = 1.0f / sum;
    for (int i = 0; i < SEQ / 32; ++i) {
      const float p = row[lane + i * 32] * inv;
      row[lane + i * 32] = p;                       // keep for A@V
      Aout[(size_t)r * SEQ + lane + i * 32] = p;    // only HBM touch of A
    }
  }
  __syncthreads();

  // ---- Phase 3: attn tile = P @ V (P from LDS, V-slice L2-resident) ------
  const __bf16* vbase = Vt + (size_t)bh * HEAD_DIM * SEQ;  // row e, stride SEQ
  const int e0 = wave * 16;  // 4 waves cover E = 64
  v8f c = {};
  for (int k0 = 0; k0 < SEQ; k0 += 32) {
    if (k0 + 32 < SEQ) {  // prefetch next K-tile of Vt (global_prefetch_b8)
      __builtin_prefetch(vbase + (size_t)(e0 + nl) * SEQ + k0 + 32, 0, 1);
    }
    v16bf af = load_a_frag_lds_f32(sc, k0);
    v16bf bm = load_b_frag_bf16(vbase, SEQ, e0, k0);
    c = __builtin_amdgcn_wmma_f32_16x16x32_bf16(false, af, false, bm,
                                                (short)0, c, false, false);
  }
  const int e = e0 + nl;
#pragma unroll
  for (int r = 0; r < 8; ++r) {
    const int l = l0 + h * 8 + r;
    attn[((size_t)b * SEQ + l) * D_MODEL + hd * HEAD_DIM + e] = (__bf16)c[r];
  }
}

// ---------------------------------------------------------------------------
// Host-side orchestration
// ---------------------------------------------------------------------------
extern "C" void kernel_launch(void* const* d_in, const int* in_sizes, int n_in,
                              void* d_out, int out_size, void* d_ws, size_t ws_size,
                              hipStream_t stream) {
  const float* queries = (const float*)d_in[0];
  const float* keys    = (const float*)d_in[1];
  const float* values  = (const float*)d_in[2];
  const float* tau     = (const float*)d_in[3];
  const float* delta   = (const float*)d_in[4];
  const float* Wq      = (const float*)d_in[5];
  const float* bq      = (const float*)d_in[6];
  const float* Wk      = (const float*)d_in[7];
  const float* bk      = (const float*)d_in[8];
  const float* Wv      = (const float*)d_in[9];
  const float* bv      = (const float*)d_in[10];
  const float* Wo      = (const float*)d_in[11];
  const float* bo      = (const float*)d_in[12];

  float* out  = (float*)d_out;
  float* Amat = out + (size_t)BATCH * SEQ * D_MODEL;  // outputs: [out | A]

  const size_t M = (size_t)BATCH * SEQ;  // 4096 rows for projection GEMMs

  // Workspace carve-up (~56 MB high-water; buffers reused once dead).
  char* w = (char*)d_ws;
  auto alloc = [&](size_t elems) -> __bf16* {
    __bf16* p = (__bf16*)w;
    w += ((elems * sizeof(__bf16)) + 255) & ~(size_t)255;
    return p;
  };
  __bf16* Xq  = alloc(M * D_MODEL);                 // queries bf16
  __bf16* Xk  = alloc(M * D_MODEL);                 // keys bf16
  __bf16* Xv  = alloc(M * D_MODEL);                 // values bf16
  __bf16* WqT = alloc((size_t)D_MODEL * D_MODEL);
  __bf16* WkT = alloc((size_t)D_MODEL * D_MODEL);
  __bf16* WvT = alloc((size_t)D_MODEL * D_MODEL);
  __bf16* WoT = alloc((size_t)D_MODEL * D_MODEL);
  __bf16* qbf = alloc(M * D_MODEL);
  __bf16* kbf = alloc(M * D_MODEL);
  __bf16* vbf = alloc(M * D_MODEL);
  __bf16* Vtr    = Xq;  // reuse: Xq dead after Q projection
  __bf16* attnbf = Xk;  // reuse: Xk dead after K projection

  // 1) Convert fp32 inputs to bf16.
  {
    const size_t n = M * D_MODEL;
    const int gs = (int)((n + 255) / 256);
    f32_to_bf16_kernel<<<gs, 256, 0, stream>>>(queries, Xq, n);
    f32_to_bf16_kernel<<<gs, 256, 0, stream>>>(keys,    Xk, n);
    f32_to_bf16_kernel<<<gs, 256, 0, stream>>>(values,  Xv, n);
  }
  // 2) Transpose + convert weights to [N][K] bf16.
  {
    const size_t n = (size_t)D_MODEL * D_MODEL;
    const int gs = (int)((n + 255) / 256);
    transpose_w_kernel<<<gs, 256, 0, stream>>>(Wq, WqT);
    transpose_w_kernel<<<gs, 256, 0, stream>>>(Wk, WkT);
    transpose_w_kernel<<<gs, 256, 0, stream>>>(Wv, WvT);
    transpose_w_kernel<<<gs, 256, 0, stream>>>(Wo, WoT);
  }
  // 3) Q/K/V projections (WMMA bf16, f32 accumulate, bf16 output).
  {
    dim3 grid(D_MODEL / 256, (unsigned)(M / 16));
    gemm_bf16_kernel<__bf16><<<grid, 128, 0, stream>>>(Xq, WqT, bq, qbf,
                                                       (int)M, D_MODEL, D_MODEL);
    gemm_bf16_kernel<__bf16><<<grid, 128, 0, stream>>>(Xk, WkT, bk, kbf,
                                                       (int)M, D_MODEL, D_MODEL);
    gemm_bf16_kernel<__bf16><<<grid, 128, 0, stream>>>(Xv, WvT, bv, vbf,
                                                       (int)M, D_MODEL, D_MODEL);
  }
  // 4) V -> [B][H][E][S] so A@V B-fragments are contiguous.
  {
    const size_t n = M * D_MODEL;
    const int gs = (int)((n + 255) / 256);
    transpose_v_kernel<<<gs, 256, 0, stream>>>(vbf, Vtr);
  }
  // 5) Fused scores + softmax + A@V (A written to HBM exactly once).
  {
    dim3 grid(SEQ / 16, BATCH * N_HEADS);
    attn_fused_kernel<<<grid, 128, 0, stream>>>(qbf, kbf, Vtr, tau, delta,
                                                Amat, attnbf);
  }
  // 6) out = attn @ Wo + bo (f32 output).
  {
    dim3 grid(D_MODEL / 256, (unsigned)(M / 16));
    gemm_bf16_kernel<float><<<grid, 128, 0, stream>>>(attnbf, WoT, bo, out,
                                                      (int)M, D_MODEL, D_MODEL);
  }
}
